// MultiHeadedAttention_51771535786524
// MI455X (gfx1250) — matrix operand order repaired and checked by reading, not compiled, not run on hardware
//
#include <hip/hip_runtime.h>
#include <hip/hip_bf16.h>
#include <math.h>

#define B_  4
#define S_  2048
#define D_  1024
#define H_  16
#define DH_ 64

typedef __attribute__((ext_vector_type(16))) _Float16 v16h;
typedef __attribute__((ext_vector_type(8)))  _Float16 v8h;
typedef __attribute__((ext_vector_type(4)))  _Float16 v4h;
typedef __attribute__((ext_vector_type(8)))  float    v8f;
typedef __attribute__((ext_vector_type(4)))  int      v4i;

__device__ __forceinline__ v8f wmma_f16(v16h a, v16h b, v8f c) {
  // D = A(16x32 f16) * B(32x16 f16) + C(16x16 f32)
  return __builtin_amdgcn_wmma_f32_16x16x32_f16(false, a, false, b, (short)0, c,
                                                false, false);
}

// A-operand (16x32 f16): lane L holds row L%16; halves 0..15 = K 2*kA..2*kA+15
// (contiguous), kA = (L>=16)?8:0.  Two aligned 16B loads.
__device__ __forceinline__ v16h load_a(const _Float16* row, int kA) {
  v8h lo = *(const v8h*)(row + 2 * kA);
  v8h hi = *(const v8h*)(row + 2 * kA + 8);
  v16h r;
#pragma unroll
  for (int i = 0; i < 8; ++i) { r[i] = lo[i]; r[i + 8] = hi[i]; }
  return r;
}

// B-operand (32x16 f16): lane L holds column L%16; halves = 16 contiguous
// K values starting at (L>=16 ? 16 : 0).  p already lane-adjusted.
__device__ __forceinline__ v16h load_b(const _Float16* p) {
  v8h lo = *(const v8h*)(p);
  v8h hi = *(const v8h*)(p + 8);
  v16h r;
#pragma unroll
  for (int i = 0; i < 8; ++i) { r[i] = lo[i]; r[i + 8] = hi[i]; }
  return r;
}

__device__ __forceinline__ v8h shfl16(v8h v) {
  union { v8h h; int i[4]; } u;
  u.h = v;
#pragma unroll
  for (int k = 0; k < 4; ++k) u.i[k] = __shfl_xor(u.i[k], 16, 32);
  return u.h;
}

// ---- CDNA5 async global->LDS copy (ASYNCcnt path), 16 bytes per lane ------
// Toolchain-probed signature: (v4i addrspace(1)*, v4i addrspace(3)*, imm, imm)
__device__ __forceinline__ void async_b128(const _Float16* gsrc,
                                           _Float16* lds_dst) {
#if __has_builtin(__builtin_amdgcn_global_load_async_to_lds_b128)
  __builtin_amdgcn_global_load_async_to_lds_b128(
      (__attribute__((address_space(1))) v4i*)(void*)gsrc,
      (__attribute__((address_space(3))) v4i*)(void*)lds_dst, 0, 0);
#else
  __attribute__((address_space(3))) _Float16* lp =
      (__attribute__((address_space(3))) _Float16*)lds_dst;
  asm volatile("global_load_async_to_lds_b128 %0, %1, off"
               :
               : "v"(lp), "v"(gsrc)
               : "memory");
#endif
}

__device__ __forceinline__ void wait_async0() {
#if __has_builtin(__builtin_amdgcn_s_wait_asynccnt)
  __builtin_amdgcn_s_wait_asynccnt(0);
#else
  asm volatile("s_wait_asynccnt 0" ::: "memory");
#endif
}

// ---------------------------------------------------------------- convert
__global__ void cvt_f32_to_f16(const float* __restrict__ src,
                               _Float16* __restrict__ dst, int n4) {
  int i = blockIdx.x * blockDim.x + threadIdx.x;
  if (i >= n4) return;
  float4 v = reinterpret_cast<const float4*>(src)[i];
  v4h o;
  o[0] = (_Float16)v.x; o[1] = (_Float16)v.y;
  o[2] = (_Float16)v.z; o[3] = (_Float16)v.w;
  reinterpret_cast<v4h*>(dst)[i] = o;
}

// ------------------------------------------------------------- GEMM core
// Block = 128 threads (4 waves) computing a 64(M) x 64(N) tile of
// Y[m,n] = sum_k A[m,k] * W[n,k].  B panel (shared by all 4 waves) is
// double-buffered in LDS via async global->LDS copies; A-operand is
// register-prefetched one k-step ahead.
// Bs row padded to 40 halves (80B) to spread LDS banks, 16B aligned.
__device__ __forceinline__ void gemm_tile_64x64(
    const _Float16* __restrict__ Arow,   // per-lane A row base
    const _Float16* __restrict__ W, int n0,
    _Float16 (*Bs)[64][40], v8f acc[4]) {
  const int tid  = threadIdx.x;
  const int lane = tid & 31;
  const int n16  = lane & 15;
  const int hh   = lane >> 4;
  const int kA   = hh * 8;
  const int cn   = tid >> 1;         // column 0..63 copied by this thread
  const int kp   = (tid & 1) * 16;   // k sub-chunk (halves)

  const _Float16* gsrc = W + (size_t)(n0 + cn) * D_ + kp;

  // prologue: stage k-chunk 0
  async_b128(gsrc, &Bs[0][cn][kp]);
  wait_async0();
  __syncthreads();

  v16h a = load_a(Arow, kA);

  for (int k0 = 0; k0 < D_; k0 += 32) {
    const int  cur  = (k0 >> 5) & 1;
    const bool more = (k0 + 32) < D_;
    v16h a_n;
    if (more) {
      async_b128(gsrc + k0 + 32, &Bs[1 - cur][cn][kp]);  // overlap w/ compute
      a_n = load_a(Arow + k0 + 32, kA);                  // reg prefetch
    }
    const _Float16* bbase = &Bs[cur][0][0];
#pragma unroll
    for (int nt = 0; nt < 4; ++nt) {
      v16h bt = load_b(bbase + (nt * 16 + n16) * 40 + hh * 16);
      acc[nt] = wmma_f16(a, bt, acc[nt]);
    }
    if (more) { wait_async0(); a = a_n; }
    __syncthreads();
  }
}

// ---------------------------------------------------------------- QKV GEMM
// blockIdx.y: 0=Q, 1=K (layout [B,H,S,DH]), 2=V (transposed [B,H,DH,S]).
__global__ __launch_bounds__(128) void gemm_qkv(
    const _Float16* __restrict__ Xh,
    const _Float16* __restrict__ Wqh, const _Float16* __restrict__ Wkh,
    const _Float16* __restrict__ Wvh,
    const float* __restrict__ bq, const float* __restrict__ bk,
    const float* __restrict__ bv,
    _Float16* __restrict__ Qh, _Float16* __restrict__ Kh,
    _Float16* __restrict__ Vt) {
  __shared__ _Float16 Bs[2][64][40];
  const int lane = threadIdx.x & 31;
  const int widx = threadIdx.x >> 5;
  const int n16  = lane & 15;
  const int hh   = lane >> 4;
  const int m0   = (blockIdx.x >> 4) * 64 + widx * 16;  // 128 M-blocks of 64
  const int n0   = (blockIdx.x & 15) * 64;              // 16 N-blocks of 64
  const int which = blockIdx.y;

  const _Float16* W   = which == 0 ? Wqh : (which == 1 ? Wkh : Wvh);
  const float*   bias = which == 0 ? bq  : (which == 1 ? bk  : bv);

  v8f acc[4] = {};
  gemm_tile_64x64(Xh + (size_t)(m0 + n16) * D_, W, n0, Bs, acc);

  const int bidx = m0 / S_;
  const int s0   = (m0 % S_) + hh * 8;
#pragma unroll
  for (int nt = 0; nt < 4; ++nt) {
    const int n  = n0 + nt * 16 + n16;
    const float bn = bias[n];
    const int h = n >> 6, dh = n & 63;
    if (which < 2) {
      _Float16* dst = (which == 0 ? Qh : Kh) +
                      ((size_t)(bidx * H_ + h) * S_) * DH_ + dh;
#pragma unroll
      for (int r = 0; r < 8; ++r)
        dst[(size_t)(s0 + r) * DH_] = (_Float16)(acc[nt][r] + bn);
    } else {
      v8h o;
#pragma unroll
      for (int r = 0; r < 8; ++r) o[r] = (_Float16)(acc[nt][r] + bn);
      *(v8h*)(Vt + ((size_t)(bidx * H_ + h) * DH_ + dh) * S_ + s0) = o;
    }
  }
}

// ---------------------------------------------------------------- attention
// One wave = (b, h, 16 queries).  Transposed flash attention:
//   scores^T(keys x queries) = K * Q^T ;  ctx^T(dh x queries) = V^T * P^T
// Per-query softmax stats live per-lane; key reductions need one shfl_xor(16).
// All independent loads (K, V, mask) are issued before the first WMMA of the
// iteration so they stay in flight during the softmax VALU work.
__global__ __launch_bounds__(128) void attn(
    const _Float16* __restrict__ Qh, const _Float16* __restrict__ Kh,
    const _Float16* __restrict__ Vt, const int* __restrict__ mask,
    _Float16* __restrict__ Ctx) {
  const int lane = threadIdx.x & 31;
  const int widx = threadIdx.x >> 5;
  const int n16  = lane & 15;
  const int hh   = lane >> 4;
  const int gw   = blockIdx.x * 4 + widx;
  const int qb = gw & 127;          // S/16 query tiles
  const int h  = (gw >> 7) & 15;
  const int b  = gw >> 11;
  const int q0 = qb * 16;

  const size_t bh = (size_t)(b * H_ + h);
  const _Float16* Qbase = Qh + bh * S_ * DH_;
  const _Float16* Kbase = Kh + bh * S_ * DH_;
  const _Float16* Vbase = Vt + bh * DH_ * S_;
  const int* mbase = mask + b * S_;

  // Q^T as B-operand, loaded once; fold 1/sqrt(DH) into Q.
  const _Float16* qrow = Qbase + (size_t)(q0 + n16) * DH_;
  v16h Bq0 = load_b(qrow + hh * 16);        // d 0..31
  v16h Bq1 = load_b(qrow + 32 + hh * 16);   // d 32..63
#pragma unroll
  for (int i = 0; i < 16; ++i) {
    Bq0[i] = (_Float16)((float)Bq0[i] * 0.125f);
    Bq1[i] = (_Float16)((float)Bq1[i] * 0.125f);
  }

  v8f acc[4] = {};
  float mstat = -1e30f, lstat = 0.f;
  const int kA = hh * 8;

  for (int kb = 0; kb < S_; kb += 32) {
    // ---- issue every independent load up front ----
    const _Float16* krow0 = Kbase + (size_t)(kb + n16) * DH_;
    const _Float16* krow1 = krow0 + 16 * DH_;
    v16h Ak00 = load_a(krow0, kA);
    v16h Ak01 = load_a(krow0 + 32, kA);
    v16h Ak10 = load_a(krow1, kA);
    v16h Ak11 = load_a(krow1 + 32, kA);
    v16h Av0  = load_a(Vbase + (size_t)( 0 + n16) * S_ + kb, kA);
    v16h Av1  = load_a(Vbase + (size_t)(16 + n16) * S_ + kb, kA);
    v16h Av2  = load_a(Vbase + (size_t)(32 + n16) * S_ + kb, kA);
    v16h Av3  = load_a(Vbase + (size_t)(48 + n16) * S_ + kb, kA);
    const v4i mk0a = *(const v4i*)(mbase + kb + hh * 8);
    const v4i mk0b = *(const v4i*)(mbase + kb + hh * 8 + 4);
    const v4i mk1a = *(const v4i*)(mbase + kb + 16 + hh * 8);
    const v4i mk1b = *(const v4i*)(mbase + kb + 16 + hh * 8 + 4);

    // ---- scores^T (rows = keys, cols = queries), already scaled via Q ----
    v8f c0 = {}, c1 = {};
    c0 = wmma_f16(Ak00, Bq0, c0);
    c0 = wmma_f16(Ak01, Bq1, c0);
    c1 = wmma_f16(Ak10, Bq0, c1);
    c1 = wmma_f16(Ak11, Bq1, c1);

    // mask (rows of c are keys: key = kb + hh*8 + r [+16 for c1])
#pragma unroll
    for (int r = 0; r < 8; ++r) {
      const int m0v = (r < 4) ? mk0a[r & 3] : mk0b[r & 3];
      const int m1v = (r < 4) ? mk1a[r & 3] : mk1b[r & 3];
      c0[r] = m0v ? c0[r] : -1e9f;
      c1[r] = m1v ? c1[r] : -1e9f;
    }

    // ---- online softmax, per query (= lane column) ----
    float rmax = c0[0];
#pragma unroll
    for (int r = 0; r < 8; ++r) rmax = fmaxf(rmax, fmaxf(c0[r], c1[r]));
    rmax = fmaxf(rmax, __shfl_xor(rmax, 16, 32));
    const float mnew = fmaxf(mstat, rmax);
    const float f = __expf(mstat - mnew);
    v8h h0, h1;
    float rsum = 0.f;
#pragma unroll
    for (int r = 0; r < 8; ++r) {
      const float p0 = __expf(c0[r] - mnew);
      const float p1 = __expf(c1[r] - mnew);
      rsum += p0 + p1;
      h0[r] = (_Float16)p0;
      h1[r] = (_Float16)p1;
    }
    rsum += __shfl_xor(rsum, 16, 32);
    lstat = lstat * f + rsum;
    mstat = mnew;
#pragma unroll
    for (int t = 0; t < 4; ++t) acc[t] *= f;

    // build P^T B-operand: lane<16 -> keys 0..15 (own c0 + partner c0 rows),
    // lane>=16 -> keys 16..31 (partner c1 + own c1 rows).
    const v8h h0p = shfl16(h0);
    const v8h h1p = shfl16(h1);
    const v8h first8  = hh ? h1p : h0;
    const v8h second8 = hh ? h1  : h0p;
    v16h Bp;
#pragma unroll
    for (int i = 0; i < 8; ++i) { Bp[i] = first8[i]; Bp[i + 8] = second8[i]; }

    // ---- ctx^T += V^T(16dh x 32keys) * P^T(32keys x 16q), 4 dh-tiles ----
    acc[0] = wmma_f16(Av0, Bp, acc[0]);
    acc[1] = wmma_f16(Av1, Bp, acc[1]);
    acc[2] = wmma_f16(Av2, Bp, acc[2]);
    acc[3] = wmma_f16(Av3, Bp, acc[3]);
  }

  const float inv = 1.0f / lstat;
  _Float16* out = Ctx + (size_t)(b * S_ + q0 + n16) * D_ + h * DH_;
#pragma unroll
  for (int t = 0; t < 4; ++t) {
    v8h o;
#pragma unroll
    for (int r = 0; r < 8; ++r) o[r] = (_Float16)(acc[t][r] * inv);
    *(v8h*)(out + t * 16 + hh * 8) = o;
  }
}

// ---------------------------------------------------------------- out GEMM
__global__ __launch_bounds__(128) void gemm_out(
    const _Float16* __restrict__ Ah, const _Float16* __restrict__ Wh,
    const float* __restrict__ bias, float* __restrict__ out) {
  __shared__ _Float16 Bs[2][64][40];
  const int lane = threadIdx.x & 31;
  const int widx = threadIdx.x >> 5;
  const int n16  = lane & 15;
  const int hh   = lane >> 4;
  const int m0   = (blockIdx.x >> 4) * 64 + widx * 16;
  const int n0   = (blockIdx.x & 15) * 64;

  v8f acc[4] = {};
  gemm_tile_64x64(Ah + (size_t)(m0 + n16) * D_, Wh, n0, Bs, acc);

#pragma unroll
  for (int nt = 0; nt < 4; ++nt) {
    const int n = n0 + nt * 16 + n16;
    const float bn = bias[n];
#pragma unroll
    for (int r = 0; r < 8; ++r)
      out[(size_t)(m0 + hh * 8 + r) * D_ + n] = acc[nt][r] + bn;
  }
}

// ---------------------------------------------------------------- launch
extern "C" void kernel_launch(void* const* d_in, const int* in_sizes, int n_in,
                              void* d_out, int out_size, void* d_ws,
                              size_t ws_size, hipStream_t stream) {
  (void)in_sizes; (void)n_in; (void)out_size; (void)ws_size;
  const float* x    = (const float*)d_in[0];
  const int*   mask = (const int*)d_in[1];
  const float* Wq   = (const float*)d_in[2];
  const float* bq   = (const float*)d_in[3];
  const float* Wk   = (const float*)d_in[4];
  const float* bk   = (const float*)d_in[5];
  const float* Wv   = (const float*)d_in[6];
  const float* bv   = (const float*)d_in[7];
  const float* Wo   = (const float*)d_in[8];
  const float* bo   = (const float*)d_in[9];
  float* out = (float*)d_out;

  const size_t nX = (size_t)B_ * S_ * D_;  // 8,388,608 elements
  const size_t nW = (size_t)D_ * D_;       // 1,048,576 elements
  _Float16* Xh  = (_Float16*)d_ws;
  _Float16* Wqh = Xh + nX;
  _Float16* Wkh = Wqh + nW;
  _Float16* Wvh = Wkh + nW;
  _Float16* Woh = Wvh + nW;
  _Float16* Qh  = Woh + nW;   // [B,H,S,DH]
  _Float16* Kh  = Qh + nX;    // [B,H,S,DH]
  _Float16* Vt  = Kh + nX;    // [B,H,DH,S]
  _Float16* Ctx = Vt + nX;    // [B*S, D]
  // total workspace: ~88 MB

  cvt_f32_to_f16<<<(unsigned)((nX / 4 + 255) / 256), 256, 0, stream>>>(x, Xh, (int)(nX / 4));
  cvt_f32_to_f16<<<(unsigned)((nW / 4 + 255) / 256), 256, 0, stream>>>(Wq, Wqh, (int)(nW / 4));
  cvt_f32_to_f16<<<(unsigned)((nW / 4 + 255) / 256), 256, 0, stream>>>(Wk, Wkh, (int)(nW / 4));
  cvt_f32_to_f16<<<(unsigned)((nW / 4 + 255) / 256), 256, 0, stream>>>(Wv, Wvh, (int)(nW / 4));
  cvt_f32_to_f16<<<(unsigned)((nW / 4 + 255) / 256), 256, 0, stream>>>(Wo, Woh, (int)(nW / 4));

  gemm_qkv<<<dim3(2048, 3), 128, 0, stream>>>(Xh, Wqh, Wkh, Wvh, bq, bk, bv,
                                              Qh, Kh, Vt);
  attn<<<2048, 128, 0, stream>>>(Qh, Kh, Vt, mask, Ctx);
  gemm_out<<<2048, 128, 0, stream>>>(Ctx, Woh, bo, out);
}
